// GwACACT_28123445854582
// MI455X (gfx1250) — matrix operand order repaired and checked by reading, not compiled, not run on hardware
//
#include <hip/hip_runtime.h>
#include <hip/hip_bf16.h>

// ---------------------------------------------------------------------------
// Problem constants (from the reference)
// ---------------------------------------------------------------------------
#define NNODES 1000
#define NPAD   1008                   // feats rows padded to a 16 multiple
#define INF    32
#define HID    128
#define MSG    128
#define DEG    16
#define PRED   2
#define OUTF   10
#define MAXM   (10 * NNODES)          // 10000 scan steps
#define NSMAX  1024                   // generous cap for num_starts
#define CAP    (NSMAX + MAXM * DEG)   // queue capacity

typedef __attribute__((ext_vector_type(16))) _Float16 v16h;
typedef __attribute__((ext_vector_type(8)))  float    v8f;

// A-fragment swizzle for a 1x256 vector: within each 32-wide K block, reorder
// groups of 8 as [0..7, 16..23, 8..15, 24..31] so that the halves needed by
// lane0 (K {0..7,16..23}) and lane16 (K {8..15,24..31}) are contiguous 32B.
__device__ __forceinline__ int xswz_idx(int k) {
    return (k & ~31) | (k & 7) | ((k & 16) >> 1) | ((k & 8) << 1);
}

// ---------------------------------------------------------------------------
// Kernel 1: init queue / accumulators
// ---------------------------------------------------------------------------
__global__ void init_state(const int* __restrict__ nsp,
                           unsigned int* __restrict__ qpack,
                           float* __restrict__ finalacc) {
    int ns  = nsp[0];
    int idx = blockIdx.x * blockDim.x + threadIdx.x;
    if (idx < NNODES * HID) finalacc[idx] = 0.0f;
    if (idx < ns && idx < NSMAX)
        qpack[idx] = (unsigned int)idx | ((unsigned int)idx << 16);
}

// ---------------------------------------------------------------------------
// Kernel 2: encoder  feats = xa @ enc_w + enc_b   (1000x32 @ 32x128)
// One wave per (16-row tile, 16-col chunk); K = 32 -> exactly one WMMA.
// feats is padded to NPAD rows so stores need no bounds checks.
// ---------------------------------------------------------------------------
__global__ __launch_bounds__(256)
void encoder_wmma(const float* __restrict__ xa, const float* __restrict__ enc_w,
                  const float* __restrict__ enc_b, float* __restrict__ feats) {
    int lane  = threadIdx.x & 31;
    int wv    = threadIdx.x >> 5;          // 0..7 -> column chunk
    int mt    = blockIdx.x;                // 16-row tile
    int arow  = mt * 16 + (lane & 15);     // A row held by this lane
    int khalf = lane >> 4;                 // 0 or 1: K sub-range select
    int col   = wv * 16 + (lane & 15);     // output column

    // A fragment (16x32 f16): lanes0-15 hold K=0..15, lanes16-31 K=16..31
    v16h a;
#pragma unroll
    for (int i = 0; i < 16; ++i) a[i] = (_Float16)0.0f;
    if (arow < NNODES) {
        const float* xr = xa + arow * INF + khalf * 16;
#pragma unroll
        for (int i = 0; i < 8; ++i) {
            a[i]     = (_Float16)xr[i];
            a[8 + i] = (_Float16)xr[8 + i];
        }
    }
    // B fragment (32x16 f16): lanes0-15 hold K=0..15, lanes16-31 K=16..31
    v16h b;
    {
        const float* wp = enc_w + (khalf * 16) * HID + col;
#pragma unroll
        for (int i = 0; i < 16; ++i) b[i] = (_Float16)wp[i * HID];
    }
    v8f c = {};
    c = __builtin_amdgcn_wmma_f32_16x16x32_f16(false, a, false, b,
                                               (short)0, c, false, false);
    // C/D layout: VGPR r -> row r (lanes0-15), row r+8 (lanes16-31)
    float bias  = enc_b[col];
    int   mbase = mt * 16 + ((lane < 16) ? 0 : 8);
#pragma unroll
    for (int r = 0; r < 8; ++r)
        feats[(mbase + r) * HID + col] = c[r] + bias;   // rows < NPAD, padded
}

// ---------------------------------------------------------------------------
// Device helper: 1x256 @ 256x128 matvec via WMMA with pre-swizzled operands.
// xswz: 256 halves in A-fragment order; Wswz: 32768 halves in B-fragment
// order (chunk-major: [wv][kb][lane][16 contiguous halves]).
// Wave wv computes output columns [wv*16, wv*16+16); result is c[0] on
// lanes 0..15 (row M=0 of the D tile).
// ---------------------------------------------------------------------------
__device__ __forceinline__ float matvec_wmma_swz(const _Float16* __restrict__ xswz,
                                                 const _Float16* __restrict__ Wswz,
                                                 int lane, int wv) {
    const bool alane = (lane == 0) || (lane == 16);
    const int  khalf = lane >> 4;
    v8f c = {};
#pragma unroll
    for (int kb = 0; kb < 8; ++kb) {       // K = 8 * 32 = 256
        v16h a;
        if (alane) {
            a = *(const v16h*)(xswz + kb * 32 + khalf * 16);   // 32B contiguous
        } else {
#pragma unroll
            for (int i = 0; i < 16; ++i) a[i] = (_Float16)0.0f;
        }
        v16h b = *(const v16h*)(Wswz + ((wv * 8 + kb) * 32 + lane) * 16);
        c = __builtin_amdgcn_wmma_f32_16x16x32_f16(false, a, false, b,
                                                   (short)0, c, false, false);
    }
    return c[0];
}

// ---------------------------------------------------------------------------
// Kernel 3: the sequential ACT message-passing scan. ONE workgroup, 256 thr.
// Weights (f16, fragment-swizzled), neighbor lists and tact all live in LDS.
// 4 barriers per processed step, 1 per skipped step.
// ---------------------------------------------------------------------------
__global__ __launch_bounds__(256)
void gnn_loop(const float* __restrict__ first_message,
              const int*   __restrict__ neighbors,
              const int*   __restrict__ nsp,
              const float* __restrict__ ns_w, const float* __restrict__ ns_b,
              const float* __restrict__ nm_w, const float* __restrict__ nm_b,
              const float* __restrict__ act_w, const float* __restrict__ act_b,
              float* __restrict__ feats, float* __restrict__ finalacc,
              unsigned int* __restrict__ qpack,
              _Float16* __restrict__ nmbuf) {
    extern __shared__ char smem[];
    _Float16* nsw_s  = (_Float16*)smem;                 // 256*128 f16, swizzled
    _Float16* nmw_s  = nsw_s + 256 * HID;               // 256*128 f16, swizzled
    _Float16* x_swz  = nmw_s + 256 * HID;               // 256 halves (A order)
    _Float16* y_swz  = x_swz + 256;                     // 256 halves (A order)
    float*    ns_f   = (float*)(y_swz + 256);           // 128
    float*    red    = ns_f + 128;                      // 8 (cross-wave partials)
    float*    actw_s = red + 8;                         // 256
    float*    nsb_s  = actw_s + 256;                    // 128
    float*    nmb_s  = nsb_s + 128;                     // 128
    int*      nbr_s  = (int*)(nmb_s + 128);             // 1000*16
    float*    tact_s = (float*)(nbr_s + NNODES * DEG);  // 1000
    int*      s_node = (int*)(tact_s + NNODES);         // [2] parity slots
    int*      s_msg  = s_node + 2;                      // [2]
    int*      s_proc = s_msg + 2;                       // [2]
    float*    s_na   = (float*)(s_proc + 2);            // [1]

    const int tid  = threadIdx.x;
    const int lane = tid & 31;
    const int wv   = tid >> 5;

    // ---- preload weights into LDS, f32 -> f16, B-fragment swizzle ----
    for (int j = tid; j < 256 * HID; j += 256) {
        int i    = j & 15;            // half within lane chunk
        int ln   = (j >> 4) & 31;     // lane
        int kb   = (j >> 9) & 7;      // K block
        int wvv  = j >> 12;           // column chunk
        int krow = kb * 32 + (ln >> 4) * 8 + (i & 7) + ((i & 8) << 1);
        int col  = wvv * 16 + (ln & 15);
        nsw_s[j] = (_Float16)ns_w[krow * HID + col];
        nmw_s[j] = (_Float16)nm_w[krow * HID + col];
    }
    actw_s[tid] = act_w[tid];
    if (tid < HID) { nsb_s[tid] = ns_b[tid]; nmb_s[tid] = nm_b[tid]; }
    for (int j = tid; j < NNODES * DEG; j += 256) nbr_s[j] = neighbors[j];
    for (int j = tid; j < NNODES; j += 256) tact_s[j] = 0.0f;
    const float actb   = act_b[0];
    const int   nstart = nsp[0];

    int head = 0, tail = nstart, nmc = 0;
    __syncthreads();

    for (int step = 0; step < MAXM; ++step) {
        if (head >= tail) break;                        // queue drained: done
        const int p = step & 1;                         // parity slot

        // ---- phase 0: scalar pop + ACT check (thread 0) ----
        float ta = 0.0f;                                // live only on t0
        if (tid == 0) {
            unsigned int q = qpack[head];
            __builtin_prefetch(&qpack[head + 1], 0, 1); // next pop
            int node  = (int)(q & 0xFFFFu);
            int msgid = (int)(q >> 16);
            ta        = tact_s[node];
            s_node[p] = node;
            s_msg[p]  = msgid;
            s_proc[p] = (ta <= 1.0f - 1e-7f) ? 1 : 0;
        }
        __syncthreads();                                // sync 1
        const int node  = s_node[p];
        const int proc  = s_proc[p];
        const int msgid = s_msg[p];
        if (!proc) { head += 1; continue; }             // halted node: pop only

        // ---- phase 1: build x = [feats[node], msg] + local ACT partial ----
        if (tid < HID) {
            float fv = feats[(size_t)node * HID + tid];
            float mv = (msgid < nstart)
                         ? first_message[(size_t)msgid * MSG + tid]
                         : (float)nmbuf[(size_t)(msgid - nstart) * MSG + tid];
            _Float16 mh = (_Float16)mv;
            x_swz[xswz_idx(tid)]       = (_Float16)fv;
            x_swz[xswz_idx(HID + tid)] = mh;
            y_swz[xswz_idx(HID + tid)] = mh;
            // ACT-gate partial: x . act_w, reduced within the wave
            float pr = fv * actw_s[tid] + mv * actw_s[HID + tid];
#pragma unroll
            for (int m = 16; m >= 1; m >>= 1) pr += __shfl_xor(pr, m, 32);
            if (lane == 0) red[wv] = pr;
        }
        __syncthreads();                                // sync 2

        if (tid == 0) {
            float dot  = red[0] + red[1] + red[2] + red[3] + actb;
            float cand = 1.0f / (1.0f + __expf(-dot));
            s_na[0]    = (ta + cand > 1.0f) ? (1.0f - ta) : cand;
        }

        // ---- phase 2: ns = relu(x @ ns_w + ns_b) via WMMA ----
        {
            float v = matvec_wmma_swz(x_swz, nsw_s, lane, wv);
            if (lane < 16) {
                int c = wv * 16 + lane;
                v += nsb_s[c];
                v  = v > 0.0f ? v : 0.0f;
                ns_f[c]            = v;
                y_swz[xswz_idx(c)] = (_Float16)v;       // state half of y
            }
        }
        __syncthreads();                                // sync 3

        // ---- phase 3: commit (overlaps the nm WMMA chain below) ----
        const float na = s_na[0];
        if (tid < HID) {
            float nv = ns_f[tid];
            feats[(size_t)node * HID + tid]     = nv;
            finalacc[(size_t)node * HID + tid] += nv * na;
        }
        if (tid < DEG) {
            unsigned int nb  = (unsigned int)nbr_s[node * DEG + tid];
            unsigned int mid = (unsigned int)(nstart + nmc);
            qpack[tail + tid] = nb | (mid << 16);
        }
        if (tid == 0) {
            tact_s[node] = ta + na;
            // prefetch next step's working set (entry is from an older,
            // already-fenced step only if head+1 < current tail)
            if (head + 1 < tail) {
                unsigned int q2 = qpack[head + 1];
                int node2  = (int)(q2 & 0xFFFFu);
                int msgid2 = (int)(q2 >> 16);
                const char* fr = (const char*)(feats + (size_t)node2 * HID);
                __builtin_prefetch(fr, 0, 1);
                __builtin_prefetch(fr + 256, 0, 1);
                if (msgid2 >= nstart) {
                    __builtin_prefetch(nmbuf + (size_t)(msgid2 - nstart) * MSG, 0, 1);
                } else {
                    const char* mr = (const char*)(first_message + (size_t)msgid2 * MSG);
                    __builtin_prefetch(mr, 0, 1);
                    __builtin_prefetch(mr + 256, 0, 1);
                }
            }
        }

        // ---- phase 4: nm = [ns,msg] @ nm_w + nm_b via WMMA, store direct ----
        {
            float v = matvec_wmma_swz(y_swz, nmw_s, lane, wv);
            if (lane < 16) {
                int c = wv * 16 + lane;
                nmbuf[(size_t)nmc * MSG + c] = (_Float16)(v + nmb_s[c]);
            }
        }

        head += 1; tail += DEG; nmc += 1;
        __threadfence_block();                          // queue + nm visible
        __syncthreads();                                // sync 4
    }
}

// ---------------------------------------------------------------------------
// Kernel 4: readout  g = sum_n final[n,:]; logits = g @ dec_w + b; log_softmax
// ---------------------------------------------------------------------------
__global__ __launch_bounds__(128)
void decode(const float* __restrict__ finalacc, const float* __restrict__ dec_w,
            const float* __restrict__ dec_b, float* __restrict__ out) {
    __shared__ float g[HID];
    __shared__ float lg[PRED * OUTF];
    int t = threadIdx.x;
    float s = 0.0f;
    for (int n = 0; n < NNODES; ++n) s += finalacc[(size_t)n * HID + t];
    g[t] = s;
    __syncthreads();
    if (t < PRED * OUTF) {
        int p = t / OUTF, o = t % OUTF;
        float acc = dec_b[p * OUTF + o];
        for (int h = 0; h < HID; ++h)
            acc += g[h] * dec_w[((size_t)p * HID + h) * OUTF + o];
        lg[t] = acc;
    }
    __syncthreads();
    if (t < PRED) {
        float mx = -1e30f;
        for (int o = 0; o < OUTF; ++o) mx = fmaxf(mx, lg[t * OUTF + o]);
        float se = 0.0f;
        for (int o = 0; o < OUTF; ++o) se += __expf(lg[t * OUTF + o] - mx);
        float lse = mx + __logf(se);
        for (int o = 0; o < OUTF; ++o) out[t * OUTF + o] = lg[t * OUTF + o] - lse;
    }
}

// ---------------------------------------------------------------------------
// Host entry
// ---------------------------------------------------------------------------
extern "C" void kernel_launch(void* const* d_in, const int* in_sizes, int n_in,
                              void* d_out, int out_size, void* d_ws, size_t ws_size,
                              hipStream_t stream) {
    (void)in_sizes; (void)n_in; (void)out_size; (void)ws_size;
    const float* xa     = (const float*)d_in[0];
    const float* fm     = (const float*)d_in[1];
    const int*   nbr    = (const int*)d_in[2];
    const int*   nsp    = (const int*)d_in[3];
    const float* enc_w  = (const float*)d_in[4];
    const float* enc_b  = (const float*)d_in[5];
    const float* ns_w   = (const float*)d_in[6];
    const float* ns_b   = (const float*)d_in[7];
    const float* nm_w   = (const float*)d_in[8];
    const float* nm_b   = (const float*)d_in[9];
    const float* act_w  = (const float*)d_in[10];
    const float* act_b  = (const float*)d_in[11];
    const float* dec_w  = (const float*)d_in[12];
    const float* dec_b  = (const float*)d_in[13];

    // carve workspace (256B aligned regions)
    char*  base = (char*)d_ws;
    size_t off  = 0;
    auto alloc = [&](size_t bytes) -> void* {
        void* p = base + off;
        off += (bytes + 255) & ~(size_t)255;
        return p;
    };
    float*        feats    = (float*)alloc((size_t)NPAD * HID * 4);   // padded
    float*        finalacc = (float*)alloc((size_t)NNODES * HID * 4);
    unsigned int* qpack    = (unsigned int*)alloc((size_t)CAP * 4);
    _Float16*     nmbuf    = (_Float16*)alloc((size_t)MAXM * MSG * 2);

    // dynamic LDS for the persistent loop
    constexpr size_t SMEM_BYTES =
        (size_t)256 * HID * 2 * 2        // ns_w + nm_w (f16, swizzled)
        + 256 * 2 + 256 * 2              // x_swz, y_swz
        + 128 * 4                        // ns_f
        + 8 * 4                          // red
        + 256 * 4                        // actw
        + 128 * 4 + 128 * 4              // biases
        + (size_t)NNODES * DEG * 4       // neighbors
        + (size_t)NNODES * 4             // tact
        + 64;                            // broadcast slots

    init_state<<<(NNODES * HID + 255) / 256, 256, 0, stream>>>(
        nsp, qpack, finalacc);
    encoder_wmma<<<(NPAD + 15) / 16, 256, 0, stream>>>(
        xa, enc_w, enc_b, feats);
    gnn_loop<<<1, 256, SMEM_BYTES, stream>>>(
        fm, nbr, nsp, ns_w, ns_b, nm_w, nm_b, act_w, act_b,
        feats, finalacc, qpack, nmbuf);
    decode<<<1, 128, 0, stream>>>(finalacc, dec_w, dec_b, (float*)d_out);
}
